// main_model_ood_70531952935005
// MI455X (gfx1250) — matrix operand order repaired and checked by reading, not compiled, not run on hardware
//
#include <hip/hip_runtime.h>

typedef __attribute__((ext_vector_type(16))) __bf16 bfrag;
typedef __attribute__((ext_vector_type(8)))  float  v8f;

#define NEG_INF (-__builtin_inff())
#define WMMA_BF16(a,b,c) __builtin_amdgcn_wmma_f32_16x16x32_bf16(false,(a),false,(b),(short)0,(c),false,false)

// ---- WMMA fragment helpers (CDNA5 16x16x32 bf16 layouts, ISA 7.12.2) ----
// A matrix 16x32 bf16, row-major [M][K] with leading dim ld.
__device__ inline bfrag load_a_frag(const __bf16* base, int ld, int row0, int k0, int lane) {
    const __bf16* p = base + (row0 + (lane & 15)) * ld + k0 + ((lane & 16) ? 8 : 0);
    bfrag a;
#pragma unroll
    for (int e = 0; e < 8; ++e) a[e] = p[e];
#pragma unroll
    for (int e = 0; e < 8; ++e) a[8 + e] = p[16 + e];
    return a;
}
// B matrix 32x16 bf16, loaded from TRANSPOSED storage BT[N][K] with leading dim ld.
__device__ inline bfrag load_b_frag(const __bf16* baseT, int ld, int col0, int k0, int lane) {
    const __bf16* p = baseT + (col0 + (lane & 15)) * ld + k0 + ((lane & 16) ? 8 : 0);
    bfrag b;
#pragma unroll
    for (int e = 0; e < 8; ++e) b[e] = p[e];
#pragma unroll
    for (int e = 0; e < 8; ++e) b[8 + e] = p[16 + e];
    return b;
}
// C/D f32 16x16: VGPR r -> row = row0 + r + (lane>=16 ? 8:0), col = col0 + lane%16
__device__ inline void store_cd(float* out, int ld, int row0, int col0, int lane, const v8f& c) {
    int r0 = row0 + ((lane & 16) ? 8 : 0);
    int cc = col0 + (lane & 15);
#pragma unroll
    for (int r = 0; r < 8; ++r) out[(r0 + r) * ld + cc] = c[r];
}

// ================= Stage 1: per-(b,d) temporal graph GCN, fully fused in LDS ==============
// grid = B*D = 1024 blocks, 256 threads (8 waves). Dynamic LDS = 172544 B.
__global__ void k_stage1(const float* __restrict__ x, const float* __restrict__ pe_inter,
                         const float* __restrict__ W_time, const float* __restrict__ b_time,
                         float* __restrict__ h_out) {
    extern __shared__ unsigned char smem[];
    __bf16* feats = (__bf16*)smem;                               // [128][288] 73728 B (x ++ pe_inter, bf16)
    float*  Sm    = (float*)(smem + 73728);                      // [128][128] 65536 B (Gram, then reused)
    __bf16* Am    = (__bf16*)(smem + 73728 + 65536);             // [128][128] 32768 B (A01 -> Anorm)
    float*  dinv  = (float*)(smem + 73728 + 65536 + 32768);      // [128]      512 B
    __bf16* WT    = (__bf16*)Sm;                                 // overlay: [64][288] Wt(d)^T bf16
    __bf16* Y1T   = (__bf16*)(((unsigned char*)Sm) + 36864);     // overlay: [64][128] (feat@W)^T bf16

    const int tid = threadIdx.x, lane = tid & 31, wave = tid >> 5;
    const int bd = blockIdx.x, d = bd & 63;
    const float* xb = x + (size_t)bd * 128 * 256;

    // load feat = [x | pe_inter] as bf16
    for (int idx = tid; idx < 128 * 288; idx += 256) {
        int n = idx / 288, c = idx % 288;
        float v = (c < 256) ? xb[n * 256 + c] : pe_inter[n * 32 + (c - 256)];
        feats[idx] = (__bf16)v;
    }
    __syncthreads();

    // S = X X^T, K = 256 (x part only). 64 tiles, 8 per wave.
    for (int t = wave; t < 64; t += 8) {
        int rt = (t >> 3) * 16, ct = (t & 7) * 16;
        v8f acc = {};
        for (int k = 0; k < 256; k += 32) {
            bfrag a = load_a_frag(feats, 288, rt, k, lane);
            bfrag b = load_b_frag(feats, 288, ct, k, lane);  // B = X^T: row-major X is already [N][K]
            acc = WMMA_BF16(a, b, acc);
        }
        store_cd(Sm, 128, rt, ct, lane, acc);
    }
    __syncthreads();

    for (int i = tid; i < 128 * 128; i += 256) Am[i] = (__bf16)0.0f;
    __syncthreads();

    // top-8 per row (self excluded), symmetrized; destructive on Sm
    if (tid < 128) {
        int i = tid;
        float* row = Sm + i * 128;
        row[i] = NEG_INF;
        for (int s = 0; s < 8; ++s) {
            float best = NEG_INF; int bj = 0;
            for (int j = 0; j < 128; ++j) { float v = row[j]; if (v > best) { best = v; bj = j; } }
            row[bj] = NEG_INF;
            Am[i * 128 + bj] = (__bf16)1.0f;
            Am[bj * 128 + i] = (__bf16)1.0f;   // benign duplicate 1.0 writes
        }
    }
    __syncthreads();
    if (tid < 128) {
        float s = 1.0f;
        for (int j = 0; j < 128; ++j) s += (float)Am[tid * 128 + j];
        dinv[tid] = rsqrtf(s);
    }
    __syncthreads();
    // Anorm = D^-1/2 (A+I) D^-1/2 in place (bf16)
    for (int idx = tid; idx < 128 * 128; idx += 256) {
        int i = idx >> 7, j = idx & 127;
        float a = (float)Am[idx] + ((i == j) ? 1.0f : 0.0f);
        Am[idx] = (__bf16)(a * dinv[i] * dinv[j]);
    }
    __syncthreads();

    // WT = W_time[d]^T (overlays dead Sm)
    const float* Wd = W_time + (size_t)d * 288 * 64;
    for (int idx = tid; idx < 64 * 288; idx += 256) {
        int f = idx / 288, c = idx % 288;
        WT[idx] = (__bf16)Wd[c * 64 + f];
    }
    __syncthreads();

    // Y1 = feat @ W_time[d]  -> store transposed Y1T[64][128]
    for (int t = wave; t < 32; t += 8) {
        int rt = (t >> 2) * 16, ct = (t & 3) * 16;
        v8f acc = {};
        for (int k = 0; k < 288; k += 32) {
            bfrag a = load_a_frag(feats, 288, rt, k, lane);
            bfrag b = load_b_frag(WT, 288, ct, k, lane);
            acc = WMMA_BF16(a, b, acc);
        }
        int r0 = rt + ((lane & 16) ? 8 : 0), cc = ct + (lane & 15);
#pragma unroll
        for (int r = 0; r < 8; ++r) Y1T[cc * 128 + (r0 + r)] = (__bf16)acc[r];
    }
    __syncthreads();

    // h = Anorm @ Y1 + b_time[d]  -> global
    const float* bt = b_time + d * 64;
    float* ho = h_out + (size_t)bd * 128 * 64;
    for (int t = wave; t < 32; t += 8) {
        int rt = (t >> 2) * 16, ct = (t & 3) * 16;
        v8f acc = {};
        for (int k = 0; k < 128; k += 32) {
            bfrag a = load_a_frag(Am, 128, rt, k, lane);
            bfrag b = load_b_frag(Y1T, 128, ct, k, lane);
            acc = WMMA_BF16(a, b, acc);
        }
        int r0 = rt + ((lane & 16) ? 8 : 0), cc = ct + (lane & 15);
#pragma unroll
        for (int r = 0; r < 8; ++r) ho[(r0 + r) * 64 + cc] = acc[r] + bt[cc];
    }
}

// ================= Stage 2: per-b dim graph: S2 = F2 F2^T (K=8208), top-4, 2-hop reach ====
__global__ void k_stage2(const float* __restrict__ h, const float* __restrict__ pe_dim,
                         float* __restrict__ Ad_out, float* __restrict__ R_out) {
    __shared__ __bf16 Cb[64 * 32];   // streamed K-chunk of feat2, bf16
    __shared__ float  S2[64 * 64];
    __shared__ float  Ad[64 * 64];
    __shared__ float  Rm[64 * 64];
    const int tid = threadIdx.x, lane = tid & 31, wave = tid >> 5;
    const int b = blockIdx.x;
    v8f acc0 = {}, acc1 = {};
    const int t0 = wave, t1 = wave + 8;   // 16 tiles (4x4), 2 per wave
    for (int c = 0; c < 257; ++c) {       // 8208 = 256*32 + 16 (tail zero-padded)
        int kbase = c * 32;
        for (int idx = tid; idx < 64 * 32; idx += 256) {
            int dim = idx >> 5, kk = kbase + (idx & 31);
            float v = 0.0f;
            if (kk < 8192)      v = h[(((size_t)b * 64 + dim) * 128 + (kk >> 6)) * 64 + (kk & 63)];
            else if (kk < 8208) v = pe_dim[dim * 16 + (kk - 8192)];
            Cb[idx] = (__bf16)v;
        }
        __syncthreads();
        {
            int rt = (t0 >> 2) * 16, ct = (t0 & 3) * 16;
            acc0 = WMMA_BF16(load_a_frag(Cb, 32, rt, 0, lane), load_b_frag(Cb, 32, ct, 0, lane), acc0);
            rt = (t1 >> 2) * 16; ct = (t1 & 3) * 16;
            acc1 = WMMA_BF16(load_a_frag(Cb, 32, rt, 0, lane), load_b_frag(Cb, 32, ct, 0, lane), acc1);
        }
        __syncthreads();
    }
    store_cd(S2, 64, (t0 >> 2) * 16, (t0 & 3) * 16, lane, acc0);
    store_cd(S2, 64, (t1 >> 2) * 16, (t1 & 3) * 16, lane, acc1);
    for (int i = tid; i < 64 * 64; i += 256) Ad[i] = 0.0f;
    __syncthreads();
    if (tid < 64) {                         // top-4, symmetrize
        int i = tid; float* row = S2 + i * 64;
        row[i] = NEG_INF;
        for (int s = 0; s < 4; ++s) {
            float best = NEG_INF; int bj = 0;
            for (int j = 0; j < 64; ++j) { float v = row[j]; if (v > best) { best = v; bj = j; } }
            row[bj] = NEG_INF;
            Ad[i * 64 + bj] = 1.0f;
            Ad[bj * 64 + i] = 1.0f;
        }
    }
    __syncthreads();
    if (tid < 64)
        for (int j = 0; j < 64; ++j) Rm[tid * 64 + j] = (tid == j) ? 1.0f : 0.0f;
    __syncthreads();
    for (int hop = 0; hop < 2; ++hop) {     // R = R | (R@A > 0)
        float nr[64];
        if (tid < 64) {
            int i = tid;
            for (int j = 0; j < 64; ++j) {
                float s = 0.0f;
                for (int k = 0; k < 64; ++k) s += Rm[i * 64 + k] * Ad[k * 64 + j];
                nr[j] = (Rm[i * 64 + j] > 0.0f || s > 0.0f) ? 1.0f : 0.0f;
            }
        }
        __syncthreads();
        if (tid < 64) for (int j = 0; j < 64; ++j) Rm[tid * 64 + j] = nr[j];
        __syncthreads();
    }
    for (int i = tid; i < 64 * 64; i += 256) {
        Ad_out[(size_t)b * 4096 + i] = Ad[i];
        R_out[(size_t)b * 4096 + i]  = Rm[i];
    }
}

// ================= Stage 3a: Y = feat2 @ W_sub (K = 8208), per b ===========================
__global__ void k_sub_y(const float* __restrict__ h, const float* __restrict__ pe_dim,
                        const float* __restrict__ W_sub, float* __restrict__ Y_out) {
    __shared__ __bf16 Fb[64 * 32];    // feat2 chunk
    __shared__ __bf16 WT[128 * 32];   // W_sub chunk transposed [n][k]
    const int tid = threadIdx.x, lane = tid & 31, wave = tid >> 5;
    const int b = blockIdx.x;
    v8f acc[4] = {};                   // 32 tiles (4x8), 4 per wave
    for (int c = 0; c < 257; ++c) {
        int kbase = c * 32;
        for (int idx = tid; idx < 64 * 32; idx += 256) {
            int dim = idx >> 5, kk = kbase + (idx & 31);
            float v = 0.0f;
            if (kk < 8192)      v = h[(((size_t)b * 64 + dim) * 128 + (kk >> 6)) * 64 + (kk & 63)];
            else if (kk < 8208) v = pe_dim[dim * 16 + (kk - 8192)];
            Fb[idx] = (__bf16)v;
        }
        for (int idx = tid; idx < 128 * 32; idx += 256) {
            int g = idx >> 5, kk = kbase + (idx & 31);
            WT[idx] = (__bf16)((kk < 8208) ? W_sub[(size_t)kk * 128 + g] : 0.0f);
        }
        __syncthreads();
#pragma unroll
        for (int q = 0; q < 4; ++q) {
            int t = wave + 8 * q;
            int rt = (t >> 3) * 16, ct = (t & 7) * 16;
            acc[q] = WMMA_BF16(load_a_frag(Fb, 32, rt, 0, lane), load_b_frag(WT, 32, ct, 0, lane), acc[q]);
        }
        __syncthreads();
    }
    float* Yo = Y_out + (size_t)b * 64 * 128;
#pragma unroll
    for (int q = 0; q < 4; ++q) {
        int t = wave + 8 * q;
        store_cd(Yo, 128, (t >> 3) * 16, (t & 7) * 16, lane, acc[q]);
    }
}

// ================= Stage 3b: per-(b,center) subgraph GCN + max pool ========================
// grid = B*64 blocks
__global__ void k_subpool(const float* __restrict__ Ad, const float* __restrict__ R,
                          const float* __restrict__ Y, const float* __restrict__ b_sub,
                          float* __restrict__ pooled) {
    __shared__ __bf16 Ab[64 * 64];    // A_d bf16 (0/1 exact)
    __shared__ __bf16 Bt[128 * 64];   // (dvec ⊙ Y)^T bf16
    __shared__ float  T[64 * 128];
    __shared__ float  Ri[64];
    __shared__ float  dvec[64];
    const int tid = threadIdx.x, lane = tid & 31, wave = tid >> 5;
    const int b = blockIdx.x >> 6, i = blockIdx.x & 63;
    const float* Ag = Ad + (size_t)b * 4096;
    const float* Yb = Y + (size_t)b * 64 * 128;
    for (int idx = tid; idx < 4096; idx += 256) Ab[idx] = (__bf16)Ag[idx];
    if (tid < 64) Ri[tid] = R[(size_t)b * 4096 + i * 64 + tid];
    __syncthreads();
    if (tid < 64) {                   // deg[i][j] = 1 + sum_k A[j][k] R[i][k]
        int j = tid; float s = 1.0f;
        for (int k = 0; k < 64; ++k) s += (float)Ab[j * 64 + k] * Ri[k];
        dvec[j] = (Ri[j] > 0.0f) ? rsqrtf(s) : 0.0f;
    }
    __syncthreads();
    for (int idx = tid; idx < 128 * 64; idx += 256) {
        int f = idx >> 6, k = idx & 63;
        Bt[idx] = (__bf16)(dvec[k] * Yb[k * 128 + f]);
    }
    __syncthreads();
    // T = A @ (dvec ⊙ Y): M=64,N=128,K=64 -> 32 tiles, 4 per wave
#pragma unroll
    for (int q = 0; q < 4; ++q) {
        int t = wave + 8 * q;
        int rt = (t >> 3) * 16, ct = (t & 7) * 16;
        v8f acc = {};
        acc = WMMA_BF16(load_a_frag(Ab, 64, rt, 0,  lane), load_b_frag(Bt, 64, ct, 0,  lane), acc);
        acc = WMMA_BF16(load_a_frag(Ab, 64, rt, 32, lane), load_b_frag(Bt, 64, ct, 32, lane), acc);
        store_cd(T, 128, rt, ct, lane, acc);
    }
    __syncthreads();
    if (tid < 128) {                  // Z + masked max pool over j
        int f = tid;
        float m = NEG_INF;
        for (int j = 0; j < 64; ++j) {
            if (Ri[j] > 0.0f) {
                float z = dvec[j] * T[j * 128 + f] + dvec[j] * dvec[j] * Yb[j * 128 + f] + b_sub[f];
                m = fmaxf(m, z);
            }
        }
        pooled[((size_t)b * 64 + i) * 128 + f] = m;
    }
}

// ================= Stage 4: dim-level GCN + global max pool ================================
// grid = B blocks, dynamic LDS = 106752 B
__global__ void k_stage4(const float* __restrict__ Ad, const float* __restrict__ pooled,
                         const float* __restrict__ W_dim, const float* __restrict__ b_dim,
                         float* __restrict__ dim_out, float* __restrict__ g_pool) {
    extern __shared__ unsigned char smem[];
    __bf16* An  = (__bf16*)smem;                                     // [64][64]   8192
    __bf16* P   = (__bf16*)(smem + 8192);                            // [64][128]  16384
    __bf16* Wt  = (__bf16*)(smem + 8192 + 16384);                    // [128][128] 32768
    __bf16* P2T = (__bf16*)(smem + 8192 + 16384 + 32768);            // [128][64]  16384
    float*  Dol = (float*)(smem + 8192 + 16384 + 32768 + 16384);     // [64][128]  32768
    float*  dv  = (float*)(smem + 8192 + 16384 + 32768 + 16384 + 32768); // [64]   256
    const int tid = threadIdx.x, lane = tid & 31, wave = tid >> 5;
    const int b = blockIdx.x;
    const float* Ag = Ad + (size_t)b * 4096;
    if (tid < 64) {
        float s = 1.0f;
        for (int j = 0; j < 64; ++j) s += Ag[tid * 64 + j];
        dv[tid] = rsqrtf(s);
    }
    __syncthreads();
    for (int idx = tid; idx < 4096; idx += 256) {
        int i = idx >> 6, j = idx & 63;
        float a = Ag[idx] + ((i == j) ? 1.0f : 0.0f);
        An[idx] = (__bf16)(a * dv[i] * dv[j]);
    }
    for (int idx = tid; idx < 64 * 128; idx += 256) P[idx] = (__bf16)pooled[(size_t)b * 8192 + idx];
    for (int idx = tid; idx < 128 * 128; idx += 256) {
        int n = idx >> 7, k = idx & 127;
        Wt[idx] = (__bf16)W_dim[k * 128 + n];
    }
    __syncthreads();
    // P2 = pooled @ W_dim (M=64,N=128,K=128) -> store transposed
#pragma unroll
    for (int q = 0; q < 4; ++q) {
        int t = wave + 8 * q;
        int rt = (t >> 3) * 16, ct = (t & 7) * 16;
        v8f acc = {};
        for (int k = 0; k < 128; k += 32)
            acc = WMMA_BF16(load_a_frag(P, 128, rt, k, lane), load_b_frag(Wt, 128, ct, k, lane), acc);
        int r0 = rt + ((lane & 16) ? 8 : 0), cc = ct + (lane & 15);
#pragma unroll
        for (int r = 0; r < 8; ++r) P2T[cc * 64 + r0 + r] = (__bf16)acc[r];
    }
    __syncthreads();
    // dim_out = Anorm @ P2 + b_dim
    float* Dg = dim_out + (size_t)b * 8192;
#pragma unroll
    for (int q = 0; q < 4; ++q) {
        int t = wave + 8 * q;
        int rt = (t >> 3) * 16, ct = (t & 7) * 16;
        v8f acc = {};
        acc = WMMA_BF16(load_a_frag(An, 64, rt, 0,  lane), load_b_frag(P2T, 64, ct, 0,  lane), acc);
        acc = WMMA_BF16(load_a_frag(An, 64, rt, 32, lane), load_b_frag(P2T, 64, ct, 32, lane), acc);
        int r0 = rt + ((lane & 16) ? 8 : 0), cc = ct + (lane & 15);
#pragma unroll
        for (int r = 0; r < 8; ++r) {
            float v = acc[r] + b_dim[cc];
            Dol[(r0 + r) * 128 + cc] = v;
            Dg[(r0 + r) * 128 + cc]  = v;
        }
    }
    __syncthreads();
    if (tid < 128) {                  // g_pool = max over D
        float m = NEG_INF;
        for (int j = 0; j < 64; ++j) m = fmaxf(m, Dol[j * 128 + tid]);
        g_pool[b * 128 + tid] = m;
    }
}

// ================= Domain head: dom = relu(dim_out@dm1w+b)@dm2w+b ==========================
// grid = B blocks (64 rows each), dynamic LDS = 81920 B
__global__ void k_dom(const float* __restrict__ dim_out,
                      const float* __restrict__ dm1w, const float* __restrict__ dm1b,
                      const float* __restrict__ dm2w, const float* __restrict__ dm2b,
                      float* __restrict__ dom) {
    extern __shared__ unsigned char smem[];
    __bf16* X1  = (__bf16*)smem;                           // [64][128]  16384
    __bf16* W1t = (__bf16*)(smem + 16384);                 // [128][128] 32768
    __bf16* Hm  = (__bf16*)(smem + 16384 + 32768);         // [64][128]  16384
    __bf16* W2t = (__bf16*)(smem + 16384 + 32768 + 16384); // [64][128]  16384
    const int tid = threadIdx.x, lane = tid & 31, wave = tid >> 5;
    const int b = blockIdx.x;
    const float* Xg = dim_out + (size_t)b * 64 * 128;
    for (int idx = tid; idx < 64 * 128; idx += 256) X1[idx] = (__bf16)Xg[idx];
    for (int idx = tid; idx < 128 * 128; idx += 256) {
        int n = idx >> 7, k = idx & 127;
        W1t[idx] = (__bf16)dm1w[k * 128 + n];
    }
    for (int idx = tid; idx < 64 * 128; idx += 256) {
        int n = idx >> 7, k = idx & 127;
        W2t[idx] = (__bf16)dm2w[k * 64 + n];
    }
    __syncthreads();
#pragma unroll
    for (int q = 0; q < 4; ++q) {      // H = relu(X@dm1w + dm1b): 32 tiles
        int t = wave + 8 * q;
        int rt = (t >> 3) * 16, ct = (t & 7) * 16;
        v8f acc = {};
        for (int k = 0; k < 128; k += 32)
            acc = WMMA_BF16(load_a_frag(X1, 128, rt, k, lane), load_b_frag(W1t, 128, ct, k, lane), acc);
        int r0 = rt + ((lane & 16) ? 8 : 0), cc = ct + (lane & 15);
#pragma unroll
        for (int r = 0; r < 8; ++r) Hm[(r0 + r) * 128 + cc] = (__bf16)fmaxf(acc[r] + dm1b[cc], 0.0f);
    }
    __syncthreads();
    float* Do = dom + (size_t)b * 64 * 64;
#pragma unroll
    for (int q = 0; q < 2; ++q) {      // dom = H@dm2w + dm2b: 16 tiles
        int t = wave + 8 * q;
        int rt = (t >> 2) * 16, ct = (t & 3) * 16;
        v8f acc = {};
        for (int k = 0; k < 128; k += 32)
            acc = WMMA_BF16(load_a_frag(Hm, 128, rt, k, lane), load_b_frag(W2t, 128, ct, k, lane), acc);
        int r0 = rt + ((lane & 16) ? 8 : 0), cc = ct + (lane & 15);
#pragma unroll
        for (int r = 0; r < 8; ++r) Do[(r0 + r) * 64 + cc] = acc[r] + dm2b[cc];
    }
}

// ================= Task head: task = relu(g_pool@c1w+b)@c2w+b ==============================
__global__ void k_task(const float* __restrict__ g_pool,
                       const float* __restrict__ c1w, const float* __restrict__ c1b,
                       const float* __restrict__ c2w, const float* __restrict__ c2b,
                       float* __restrict__ task) {
    __shared__ __bf16 Gp[16 * 128];
    __shared__ __bf16 W1t[128 * 128];
    __shared__ float  H1[16 * 128];
    const int tid = threadIdx.x, lane = tid & 31, wave = tid >> 5;
    for (int idx = tid; idx < 16 * 128; idx += 256) Gp[idx] = (__bf16)g_pool[idx];
    for (int idx = tid; idx < 128 * 128; idx += 256) {
        int n = idx >> 7, k = idx & 127;
        W1t[idx] = (__bf16)c1w[k * 128 + n];
    }
    __syncthreads();
    {   // M=16 (1 row tile), N=128 (8 tiles): one tile per wave
        int ct = wave * 16;
        v8f acc = {};
        for (int k = 0; k < 128; k += 32)
            acc = WMMA_BF16(load_a_frag(Gp, 128, 0, k, lane), load_b_frag(W1t, 128, ct, k, lane), acc);
        int r0 = ((lane & 16) ? 8 : 0), cc = ct + (lane & 15);
#pragma unroll
        for (int r = 0; r < 8; ++r) H1[(r0 + r) * 128 + cc] = fmaxf(acc[r] + c1b[cc], 0.0f);
    }
    __syncthreads();
    if (tid < 32) {                    // [16,128]@[128,2]
        int i = tid >> 1, c = tid & 1;
        float s = c2b[c];
        for (int k = 0; k < 128; ++k) s += H1[i * 128 + k] * c2w[k * 2 + c];
        task[i * 2 + c] = s;
    }
}

extern "C" void kernel_launch(void* const* d_in, const int* in_sizes, int n_in,
                              void* d_out, int out_size, void* d_ws, size_t ws_size,
                              hipStream_t stream) {
    (void)in_sizes; (void)n_in; (void)out_size; (void)ws_size;
    const float* x        = (const float*)d_in[0];
    const float* pe_inter = (const float*)d_in[1];
    const float* pe_dim   = (const float*)d_in[2];
    const float* W_time   = (const float*)d_in[3];
    const float* b_time   = (const float*)d_in[4];
    const float* W_sub    = (const float*)d_in[5];
    const float* b_sub    = (const float*)d_in[6];
    const float* W_dim    = (const float*)d_in[7];
    const float* b_dim    = (const float*)d_in[8];
    const float* c1w      = (const float*)d_in[9];
    const float* c1b      = (const float*)d_in[10];
    const float* c2w      = (const float*)d_in[11];
    const float* c2b      = (const float*)d_in[12];
    const float* dm1w     = (const float*)d_in[13];
    const float* dm1b     = (const float*)d_in[14];
    const float* dm2w     = (const float*)d_in[15];
    const float* dm2b     = (const float*)d_in[16];

    // workspace layout (floats): h | A_d | R | Y | pooled | dim_out  (~35.7 MB)
    float* ws      = (float*)d_ws;
    float* h       = ws;                    // 16*64*128*64 = 8,388,608
    float* Adw     = h + 8388608;           // 16*64*64     = 65,536
    float* Rw      = Adw + 65536;           // 65,536
    float* Yw      = Rw + 65536;            // 16*64*128    = 131,072
    float* pooledw = Yw + 131072;           // 131,072
    float* dimoutw = pooledw + 131072;      // 131,072

    // output layout: task [16,2] | dom [1024,64] | g_pool [16,128]
    float* out       = (float*)d_out;
    float* task_out  = out;
    float* dom_out   = out + 32;
    float* gpool_out = out + 32 + 65536;

    // large-LDS kernels need the attribute bumped (host-side, not stream-ordered)
    hipFuncSetAttribute((const void*)k_stage1, hipFuncAttributeMaxDynamicSharedMemorySize, 172544);
    hipFuncSetAttribute((const void*)k_stage4, hipFuncAttributeMaxDynamicSharedMemorySize, 106752);
    hipFuncSetAttribute((const void*)k_dom,    hipFuncAttributeMaxDynamicSharedMemorySize, 81920);

    k_stage1 <<<dim3(1024), dim3(256), 172544, stream>>>(x, pe_inter, W_time, b_time, h);
    k_stage2 <<<dim3(16),   dim3(256), 0,      stream>>>(h, pe_dim, Adw, Rw);
    k_sub_y  <<<dim3(16),   dim3(256), 0,      stream>>>(h, pe_dim, W_sub, Yw);
    k_subpool<<<dim3(1024), dim3(256), 0,      stream>>>(Adw, Rw, Yw, b_sub, pooledw);
    k_stage4 <<<dim3(16),   dim3(256), 106752, stream>>>(Adw, pooledw, W_dim, b_dim, dimoutw, gpool_out);
    k_dom    <<<dim3(16),   dim3(256), 81920,  stream>>>(dimoutw, dm1w, dm1b, dm2w, dm2b, dom_out);
    k_task   <<<dim3(1),    dim3(256), 0,      stream>>>(gpool_out, c1w, c1b, c2w, c2b, task_out);
}
